// NONLocalBlock3D_13761075216909
// MI455X (gfx1250) — matrix operand order repaired and checked by reading, not compiled, not run on hardware
//
#include <hip/hip_runtime.h>

typedef __attribute__((ext_vector_type(16))) _Float16 v16h;
typedef __attribute__((ext_vector_type(4)))  _Float16 h4;
typedef __attribute__((ext_vector_type(8)))  float    v8f;

#define B_   8
#define C_   256
#define CI_  128
#define N_   6272     // T*H*W = 8*28*28
#define NK_  1568     // 8*14*14
#define EPSBN 1e-5f

union FragU { v16h v; uint4 q[2]; };

// A-matrix fragment (16x32 f16). Per ISA: lane l=lane&15 holds row M=l.
// hi=lane>>4: j<8 -> K = 8*hi + j ; j>=8 -> K = 16 + 8*hi + (j-8).
// Memory row-major [M][K], row stride ld (halves): two 16B loads.
__device__ __forceinline__ v16h load_a(const _Float16* base, int ld, int lane) {
  const int l = lane & 15, h = lane >> 4;
  const _Float16* p = base + (size_t)l * ld + 8 * h;
  FragU f;
  f.q[0] = *(const uint4*)(p);
  f.q[1] = *(const uint4*)(p + 16);
  return f.v;
}

// B-matrix fragment (32x16 f16). Same K order as A: hi=lane>>4: j<8 -> K = 8*hi + j ;
// j>=8 -> K = 16 + 8*hi + (j-8), column N = lane&15. Memory row-major [N][K] (B stored
// transposed, K contiguous): two 16B loads.
__device__ __forceinline__ v16h load_b(const _Float16* base, int ld, int lane) {
  const int l = lane & 15, h = lane >> 4;
  const _Float16* p = base + (size_t)l * ld + 8 * h;
  FragU f;
  f.q[0] = *(const uint4*)(p);
  f.q[1] = *(const uint4*)(p + 16);
  return f.v;
}

__device__ __forceinline__ v8f wmma_f16(v16h a, v16h b, v8f c) {
  return __builtin_amdgcn_wmma_f32_16x16x32_f16(false, a, false, b, (short)0, c,
                                                false, false);
}

__device__ __forceinline__ v8f vzero8() {
  v8f z;
#pragma unroll
  for (int i = 0; i < 8; ++i) z[i] = 0.0f;
  return z;
}

__device__ __forceinline__ void wait_ds0() {
  asm volatile("s_wait_dscnt 0" ::: "memory");
}

__device__ __forceinline__ void sched_fence() {
#if __has_builtin(__builtin_amdgcn_sched_barrier)
  __builtin_amdgcn_sched_barrier(0);
#endif
}

// ---------------------------------------------------------------------------
// Kernel 0: convert output-projection weight to f16, zero BN stat accumulators
// ---------------------------------------------------------------------------
__global__ __launch_bounds__(256) void init_kernel(const float* __restrict__ Ww,
                                                   _Float16* __restrict__ Ww16,
                                                   float* __restrict__ stats) {
  const int i = blockIdx.x * 256 + threadIdx.x;
  if (i < C_ * CI_) Ww16[i] = (_Float16)Ww[i];
  if (i < 512) stats[i] = 0.0f;
}

// ---------------------------------------------------------------------------
// Kernel 1: fused 1x1x1 projections  out[n,o] = sum_c x[c,n]*w[o,c] + b[o]
// x is channel-major -> stage transposed f16 tiles in LDS, WMMA f16 16x16x32.
// grid: (N/128, B, 3 {theta,phi,g}); block 256 (8 waves, each owns 16 rows).
// ---------------------------------------------------------------------------
__global__ __launch_bounds__(256, 2) void proj_kernel(
    const float* __restrict__ x,
    const float* __restrict__ wT, const float* __restrict__ wP,
    const float* __restrict__ wG,
    const float* __restrict__ bT, const float* __restrict__ bP,
    const float* __restrict__ bG,
    _Float16* __restrict__ oT, _Float16* __restrict__ oP,
    _Float16* __restrict__ oG) {
  __shared__ _Float16 At[128][40];  // [n][c-chunk], padded
  __shared__ _Float16 Wt[128][40];  // [o][c-chunk], padded

  const int b  = blockIdx.y;
  const int n0 = blockIdx.x * 128;
  const float* w; const float* bias; _Float16* out;
  if (blockIdx.z == 0)      { w = wT; bias = bT; out = oT; }
  else if (blockIdx.z == 1) { w = wP; bias = bP; out = oP; }
  else                      { w = wG; bias = bG; out = oG; }

  const int tid = threadIdx.x, lane = tid & 31, wave = tid >> 5;
  const int l = lane & 15, hi = lane >> 4;
  const int m0 = wave * 16;

  v8f acc[8];
#pragma unroll
  for (int oc = 0; oc < 8; ++oc) acc[oc] = vzero8();

  for (int kc = 0; kc < 8; ++kc) {
    const int c0 = kc * 32;
    {  // stage x[c0..c0+31][n0..n0+127] transposed into At[n][c]
      const int crow = tid >> 3;         // 0..31
      const int nb   = (tid & 7) * 16;   // 0..112
      const float* xp = x + ((size_t)(b * C_ + c0 + crow)) * N_ + n0 + nb;
#pragma unroll
      for (int i = 0; i < 16; i += 4) {
        const float4 v = *(const float4*)(xp + i);
        At[nb + i + 0][crow] = (_Float16)v.x;
        At[nb + i + 1][crow] = (_Float16)v.y;
        At[nb + i + 2][crow] = (_Float16)v.z;
        At[nb + i + 3][crow] = (_Float16)v.w;
      }
    }
    {  // stage w[0..127][c0..c0+31] into Wt[o][c] with packed 8B stores
      const int o  = tid >> 1;
      const int cb = (tid & 1) * 16;
      const float* wp = w + (size_t)o * C_ + c0 + cb;
#pragma unroll
      for (int i = 0; i < 16; i += 4) {
        const float4 v = *(const float4*)(wp + i);
        const h4 hv = {(_Float16)v.x, (_Float16)v.y, (_Float16)v.z,
                       (_Float16)v.w};
        *(h4*)&Wt[o][cb + i] = hv;
      }
    }
    __syncthreads();
    // preload all fragments, then issue back-to-back WMMAs
    const v16h a = load_a(&At[m0][0], 40, lane);
    v16h bb[8];
#pragma unroll
    for (int oc = 0; oc < 8; ++oc) bb[oc] = load_b(&Wt[oc * 16][0], 40, lane);
    sched_fence();
#pragma unroll
    for (int oc = 0; oc < 8; ++oc) acc[oc] = wmma_f16(a, bb[oc], acc[oc]);
    sched_fence();
    __syncthreads();
  }

#pragma unroll
  for (int oc = 0; oc < 8; ++oc) {
    const int o = oc * 16 + l;
    const float bv = bias[o];
#pragma unroll
    for (int r = 0; r < 8; ++r) {
      const int n = n0 + m0 + r + 8 * hi;
      out[((size_t)b * N_ + n) * CI_ + o] = (_Float16)(acc[oc][r] + bv);
    }
  }
}

// ---------------------------------------------------------------------------
// Kernel 2: 2x2 spatial max-pool of phi_full/g_full.
// phi -> [b][k][c] (K contiguous-in-c for QK^T B frags)
// g   -> [b][c][k] transposed (K contiguous-in-k for PV B frags)
// ---------------------------------------------------------------------------
__global__ __launch_bounds__(256) void pool_kernel(
    const _Float16* __restrict__ pf, const _Float16* __restrict__ gf,
    _Float16* __restrict__ phi, _Float16* __restrict__ gt) {
  const int i = blockIdx.x * 256 + threadIdx.x;
  if (i >= B_ * NK_ * CI_) return;
  const int c = i & (CI_ - 1);
  const int k = (i >> 7) % NK_;
  const int b = i / (NK_ * CI_);
  const int t  = k / 196;
  const int hw = k % 196;
  const int ho = hw / 14, wo = hw % 14;
  const int nb = t * 784 + (2 * ho) * 28 + 2 * wo;
  const size_t base = ((size_t)b * N_) * CI_ + c;
  const size_t i00 = base + (size_t)nb * CI_;
  const size_t i01 = i00 + CI_;
  const size_t i10 = i00 + 28 * CI_;
  const size_t i11 = i10 + CI_;
  const float p = fmaxf(fmaxf((float)pf[i00], (float)pf[i01]),
                        fmaxf((float)pf[i10], (float)pf[i11]));
  const float g = fmaxf(fmaxf((float)gf[i00], (float)gf[i01]),
                        fmaxf((float)gf[i10], (float)gf[i11]));
  phi[((size_t)b * NK_ + k) * CI_ + c] = (_Float16)p;
  gt [((size_t)b * CI_ + c) * NK_ + k] = (_Float16)g;
}

// ---------------------------------------------------------------------------
// Kernel 3: flash attention.  y[q,:] = softmax(theta[q]·phiᵀ) · gᵀ
// grid (N/128, B); block 256 = 8 waves; each wave owns 16 query rows,
// streams 32 keys per step (2 WMMA S tiles), online softmax, PV via WMMA.
// ---------------------------------------------------------------------------
__global__ __launch_bounds__(256, 2) void attn_kernel(
    const _Float16* __restrict__ theta, const _Float16* __restrict__ phi,
    const _Float16* __restrict__ gt, _Float16* __restrict__ y) {
  __shared__ _Float16 Pld[8][16][32];  // per-wave D-layout -> A-layout scratch

  const int b = blockIdx.y;
  const int tid = threadIdx.x, lane = tid & 31, wave = tid >> 5;
  const int l = lane & 15, hi = lane >> 4;
  const int q0 = blockIdx.x * 128 + wave * 16;

  const _Float16* th = theta + ((size_t)b * N_ + q0) * CI_;
  v16h Aq[4];
#pragma unroll
  for (int cc = 0; cc < 4; ++cc) Aq[cc] = load_a(th + cc * 32, CI_, lane);

  v8f O[8];
  float Mx[8], L[8];
#pragma unroll
  for (int oc = 0; oc < 8; ++oc) O[oc] = vzero8();
#pragma unroll
  for (int r = 0; r < 8; ++r) { Mx[r] = -3.0e38f; L[r] = 0.0f; }

  _Float16* Pw = &Pld[wave][0][0];
  const _Float16* gbase = gt + (size_t)b * CI_ * NK_;

  for (int kt = 0; kt < NK_ / 32; ++kt) {
    const int k0 = kt * 32;
    const _Float16* ph = phi + ((size_t)b * NK_ + k0) * CI_;
    __builtin_prefetch(ph + 32 * CI_, 0, 1);           // next K tile
    __builtin_prefetch(gbase + k0 + 32, 0, 1);         // next V tile

    // ---- S = Q·K^T : preload 8 key fragments, then 8 back-to-back WMMAs ----
    v16h Bk[8];
#pragma unroll
    for (int cc = 0; cc < 4; ++cc) {
      Bk[cc]     = load_b(ph + cc * 32, CI_, lane);
      Bk[4 + cc] = load_b(ph + 16 * CI_ + cc * 32, CI_, lane);
    }
    sched_fence();
    v8f S0 = vzero8(), S1 = vzero8();
#pragma unroll
    for (int cc = 0; cc < 4; ++cc) {
      S0 = wmma_f16(Aq[cc], Bk[cc], S0);
      S1 = wmma_f16(Aq[cc], Bk[4 + cc], S1);
    }
    sched_fence();

    // ---- hoist the V-tile fragment loads so softmax VALU hides latency ----
    const _Float16* gp = gbase + k0;
    v16h Bg[8];
#pragma unroll
    for (int oc = 0; oc < 8; ++oc)
      Bg[oc] = load_b(gp + (size_t)(oc * 16) * NK_, NK_, lane);

    // ---- online softmax (row = r + 8*hi, 16 cols across 16-lane half) ----
#pragma unroll
    for (int r = 0; r < 8; ++r) {
      float mx = fmaxf(S0[r], S1[r]);
#pragma unroll
      for (int off = 8; off >= 1; off >>= 1)
        mx = fmaxf(mx, __shfl_xor(mx, off, 32));
      const float mnew  = fmaxf(Mx[r], mx);
      const float alpha = __expf(Mx[r] - mnew);
      const float p0 = __expf(S0[r] - mnew);
      const float p1 = __expf(S1[r] - mnew);
      float s = p0 + p1;
#pragma unroll
      for (int off = 8; off >= 1; off >>= 1) s += __shfl_xor(s, off, 32);
      L[r]  = L[r] * alpha + s;
      Mx[r] = mnew;
#pragma unroll
      for (int oc = 0; oc < 8; ++oc) O[oc][r] *= alpha;
      Pw[(r + 8 * hi) * 32 + l]      = (_Float16)p0;
      Pw[(r + 8 * hi) * 32 + 16 + l] = (_Float16)p1;
    }
    wait_ds0();  // cross-lane LDS RAW within the wave

    const v16h Pa = load_a(Pw, 32, lane);
    sched_fence();
#pragma unroll
    for (int oc = 0; oc < 8; ++oc) O[oc] = wmma_f16(Pa, Bg[oc], O[oc]);
    sched_fence();
  }

  float rinv[8];
#pragma unroll
  for (int r = 0; r < 8; ++r) rinv[r] = 1.0f / L[r];
#pragma unroll
  for (int oc = 0; oc < 8; ++oc) {
#pragma unroll
    for (int r = 0; r < 8; ++r) {
      const int n = q0 + r + 8 * hi;
      y[((size_t)b * N_ + n) * CI_ + oc * 16 + l] =
          (_Float16)(O[oc][r] * rinv[r]);
    }
  }
}

// ---------------------------------------------------------------------------
// Kernel 4: output projection Wy = y·Wwᵀ + Wb (WMMA) + per-channel BN stats
// grid (N/64, B); block 256 = 8 waves: (row-group 0..3) x (o-half 0..1).
// ---------------------------------------------------------------------------
__global__ __launch_bounds__(256, 2) void wproj_kernel(
    const _Float16* __restrict__ y, const _Float16* __restrict__ Ww16,
    const float* __restrict__ Wb, float* __restrict__ wy,
    float* __restrict__ gsum, float* __restrict__ gsq) {
  __shared__ float bs[C_], bq[C_];
  const int tid = threadIdx.x, lane = tid & 31, wave = tid >> 5;
  const int l = lane & 15, hi = lane >> 4;
  if (tid < C_) { bs[tid] = 0.0f; bq[tid] = 0.0f; }
  __syncthreads();

  const int b  = blockIdx.y;
  const int n0 = blockIdx.x * 64 + (wave >> 1) * 16;
  const int o0 = (wave & 1) * 128;

  const _Float16* yp = y + ((size_t)b * N_ + n0) * CI_;
  v16h Ay[4];
#pragma unroll
  for (int cc = 0; cc < 4; ++cc) Ay[cc] = load_a(yp + cc * 32, CI_, lane);

  v8f acc[8];
#pragma unroll
  for (int oc = 0; oc < 8; ++oc) acc[oc] = vzero8();
#pragma unroll
  for (int cc = 0; cc < 4; ++cc) {
    v16h bb[8];
#pragma unroll
    for (int oc = 0; oc < 8; ++oc)
      bb[oc] = load_b(Ww16 + (size_t)(o0 + oc * 16) * CI_ + cc * 32, CI_, lane);
    sched_fence();
#pragma unroll
    for (int oc = 0; oc < 8; ++oc) acc[oc] = wmma_f16(Ay[cc], bb[oc], acc[oc]);
    sched_fence();
  }

#pragma unroll
  for (int oc = 0; oc < 8; ++oc) {
    const int o = o0 + oc * 16 + l;
    const float bv = Wb[o];
    float s1 = 0.0f, s2 = 0.0f;
#pragma unroll
    for (int r = 0; r < 8; ++r) {
      const float v = acc[oc][r] + bv;
      const int n = n0 + r + 8 * hi;
      wy[((size_t)b * C_ + o) * N_ + n] = v;
      s1 += v;
      s2 += v * v;
    }
    s1 += __shfl_xor(s1, 16, 32);  // fold the two row-halves (same o)
    s2 += __shfl_xor(s2, 16, 32);
    if (hi == 0) { atomicAdd(&bs[o], s1); atomicAdd(&bq[o], s2); }
  }
  __syncthreads();
  if (tid < C_) { atomicAdd(&gsum[tid], bs[tid]); atomicAdd(&gsq[tid], bq[tid]); }
}

// ---------------------------------------------------------------------------
// Kernel 5: BatchNorm (batch stats) + affine + residual, vectorized float4.
// ---------------------------------------------------------------------------
__global__ __launch_bounds__(256) void bn_kernel(
    const float* __restrict__ wy, const float* __restrict__ x,
    const float* __restrict__ gsum, const float* __restrict__ gsq,
    const float* __restrict__ gamma, const float* __restrict__ beta,
    float* __restrict__ out) {
  const size_t i4 = ((size_t)blockIdx.x * 256 + threadIdx.x) * 4;
  if (i4 >= (size_t)B_ * C_ * N_) return;
  const int o = (int)((i4 / N_) % C_);
  const float cnt  = (float)(B_ * N_);
  const float mean = gsum[o] / cnt;
  const float var  = gsq[o] / cnt - mean * mean;
  const float sc   = gamma[o] * rsqrtf(var + EPSBN);
  const float sh   = beta[o] - mean * sc;
  const float4 wv = *(const float4*)(wy + i4);
  const float4 xv = *(const float4*)(x + i4);
  float4 ov;
  ov.x = wv.x * sc + sh + xv.x;
  ov.y = wv.y * sc + sh + xv.y;
  ov.z = wv.z * sc + sh + xv.z;
  ov.w = wv.w * sc + sh + xv.w;
  *(float4*)(out + i4) = ov;
}

// ---------------------------------------------------------------------------
extern "C" void kernel_launch(void* const* d_in, const int* in_sizes, int n_in,
                              void* d_out, int out_size, void* d_ws,
                              size_t ws_size, hipStream_t stream) {
  (void)in_sizes; (void)n_in; (void)out_size; (void)ws_size;
  const float* x     = (const float*)d_in[0];
  const float* g_w   = (const float*)d_in[1];
  const float* g_b   = (const float*)d_in[2];
  const float* th_w  = (const float*)d_in[3];
  const float* th_b  = (const float*)d_in[4];
  const float* ph_w  = (const float*)d_in[5];
  const float* ph_b  = (const float*)d_in[6];
  const float* W_w   = (const float*)d_in[7];
  const float* W_b   = (const float*)d_in[8];
  const float* gamma = (const float*)d_in[9];
  const float* beta  = (const float*)d_in[10];
  float* out = (float*)d_out;

  char* ws = (char*)d_ws;
  size_t off = 0;
  auto alloc = [&](size_t bytes) -> char* {
    off = (off + 255) & ~(size_t)255;
    char* p = ws + off;
    off += bytes;
    return p;
  };
  _Float16* theta16 = (_Float16*)alloc((size_t)B_ * N_  * CI_ * 2);
  _Float16* phifull = (_Float16*)alloc((size_t)B_ * N_  * CI_ * 2);
  _Float16* gfull   = (_Float16*)alloc((size_t)B_ * N_  * CI_ * 2);
  _Float16* phi16   = (_Float16*)alloc((size_t)B_ * NK_ * CI_ * 2);
  _Float16* gt16    = (_Float16*)alloc((size_t)B_ * CI_ * NK_ * 2);
  _Float16* y16     = (_Float16*)alloc((size_t)B_ * N_  * CI_ * 2);
  _Float16* Ww16    = (_Float16*)alloc((size_t)C_ * CI_ * 2);
  float*    stats   = (float*)   alloc(512 * 4);
  float*    wy      = (float*)   alloc((size_t)B_ * C_ * N_ * 4);
  float* gsum = stats;
  float* gsq  = stats + 256;

  init_kernel<<<dim3(128), dim3(256), 0, stream>>>(W_w, Ww16, stats);
  proj_kernel<<<dim3(N_ / 128, B_, 3), dim3(256), 0, stream>>>(
      x, th_w, ph_w, g_w, th_b, ph_b, g_b, theta16, phifull, gfull);
  pool_kernel<<<dim3((B_ * NK_ * CI_) / 256), dim3(256), 0, stream>>>(
      phifull, gfull, phi16, gt16);
  attn_kernel<<<dim3(N_ / 128, B_), dim3(256), 0, stream>>>(theta16, phi16,
                                                            gt16, y16);
  wproj_kernel<<<dim3(N_ / 64, B_), dim3(256), 0, stream>>>(y16, Ww16, W_b, wy,
                                                            gsum, gsq);
  bn_kernel<<<dim3((int)(((size_t)B_ * C_ * N_ / 4) / 256)), dim3(256), 0,
              stream>>>(wy, x, gsum, gsq, gamma, beta, out);
}
